// LSTM_82257213653440
// MI455X (gfx1250) — matrix operand order repaired and checked by reading, not compiled, not run on hardware
//
#include <hip/hip_runtime.h>
#include <hip/hip_bf16.h>

typedef __attribute__((ext_vector_type(16))) __bf16       v16bf;
typedef __attribute__((ext_vector_type(8)))  float        v8f;
typedef __attribute__((ext_vector_type(4)))  unsigned int u32x4;

typedef const __attribute__((address_space(1))) u32x4*  gcu32x4p;
typedef const __attribute__((address_space(1))) float*  gcfloatp;
typedef       __attribute__((address_space(1))) float*  gfloatp;

#define T_STEPS 512
#define BATCH   256
#define NPTS    34
#define NH      512
#define NG      2048            // 4*NH gate columns
#define KTOT    576             // 512 (h) + 34 (x) + 30 (zero pad), 18 chunks of 32
#define KCH     18
#define NTGT    16

#define WC_ELEMS   (NG * KTOT)          // packed combined weight, bf16
#define WL_ELEMS   (NTGT * NH)          // packed W_lin, bf16
#define WC_BYTES   (WC_ELEMS * 2)
#define WL_BYTES   (WL_ELEMS * 2)
#define TILE_BYTES (KCH * 1024)         // bytes per packed N-tile (18 chunks x 1KB)

union FragU { u32x4 u[2]; v16bf v; };

// Load a 16x32 bf16 B/A fragment (32 B/lane) from global memory by integer
// address, through explicit address_space(1) so codegen uses global_load_b128.
__device__ __forceinline__ v16bf ld_frag_gaddr(unsigned long long a) {
  FragU f;
  f.u[0] = *(gcu32x4p)(a);
  f.u[1] = *(gcu32x4p)(a + 16);
  return f.v;
}

// ---------------------------------------------------------------------------
// Prep kernel 1: pack Wc = [W_hh | W_ih | 0] (2048 x 576) f32 -> bf16 in the
// wave32 B-fragment layout used here for V_WMMA_*_16X16X32_BF16 (contiguous K, as the A loads):
//   element e of lane l in tile (n_tile, k_chunk):
//     col N = n_tile*16 + (l%16),  K = k_chunk*32 + (l/16)*16 + e
//   linear dst = ((n_tile*18 + k_chunk)*32 + lane)*16 + e
// ---------------------------------------------------------------------------
__global__ void pack_wc_kernel(const float* __restrict__ W_ih,
                               const float* __restrict__ W_hh,
                               unsigned short* __restrict__ WcRaw) {
  __bf16* Wc = (__bf16*)WcRaw;
  int idx = blockIdx.x * blockDim.x + threadIdx.x;
  if (idx >= WC_ELEMS) return;
  int e    = idx & 15;
  int lane = (idx >> 4) & 31;
  int c    = (idx >> 9) % KCH;
  int tile = idx / (512 * KCH);
  int g = tile * 16 + (lane & 15);
  int k = c * 32 + ((lane >> 4) << 4) + e;
  float v = 0.0f;
  if (k < NH)             v = W_hh[(size_t)g * NH + k];
  else if (k < NH + NPTS) v = W_ih[(size_t)g * NPTS + (k - NH)];
  Wc[idx] = (__bf16)v;
}

// Prep kernel 2: pack W_lin (16 x 512): one N-tile x 16 K-chunks, same layout.
__global__ void pack_wlin_kernel(const float* __restrict__ W_lin,
                                 unsigned short* __restrict__ WlRaw) {
  __bf16* Wl = (__bf16*)WlRaw;
  int idx = blockIdx.x * blockDim.x + threadIdx.x;
  if (idx >= WL_ELEMS) return;
  int e    = idx & 15;
  int lane = (idx >> 4) & 31;
  int c    = idx >> 9;                       // 0..15
  int tgt  = lane & 15;
  int k    = c * 32 + ((lane >> 4) << 4) + e;
  Wl[idx] = (__bf16)W_lin[(size_t)tgt * NH + k];
}

// Prep kernel 3: bias_comb = b_ih + b_hh
__global__ void bias_kernel(const float* __restrict__ b_ih,
                            const float* __restrict__ b_hh,
                            float* __restrict__ bc) {
  int g = blockIdx.x * blockDim.x + threadIdx.x;
  if (g < NG) bc[g] = b_ih[g] + b_hh[g];
}

// ---------------------------------------------------------------------------
// Persistent LSTM kernel: 16 blocks x 512 threads (16 waves).  Block b owns
// batch rows [16b, 16b+16).  Wave w owns hidden slice [32w, 32w+32) and gate
// column tiles {g*32 + 2w + q}.  c-state in registers, h-state (bf16) in LDS,
// Wc re-streamed from L2 every step.  The weight base address is laundered
// per timestep (defeats LICM -> no hoist+spill) but loads go through explicit
// address_space(1) pointers (global_load_b128, not flat).
// ---------------------------------------------------------------------------
__global__ __launch_bounds__(512, 1)
void lstm_persist_kernel(const float* __restrict__ points,
                         const unsigned short* __restrict__ WcRaw,
                         const unsigned short* __restrict__ WlRaw,
                         const float* __restrict__ bias_c,
                         const float* __restrict__ b_lin,
                         float* __restrict__ out) {
  __shared__ __bf16 h_lds[16][KTOT];     // [batch row][K col]  (h | x | pad)
  __shared__ float  lpart[16 * 256];     // per-wave partial logits tiles
  __shared__ float  lred[256];           // reduced logits 16x16
  __shared__ float  blin_s[NTGT];

  const int tid   = threadIdx.x;
  const int lane  = tid & 31;
  const int w     = __builtin_amdgcn_readfirstlane(tid >> 5);  // wave id, SGPR
  const int grp   = lane >> 4;           // lane half (K-half selector)
  const int ln16  = lane & 15;
  const int bbase = blockIdx.x * 16;

  const unsigned long long points_a = (unsigned long long)points;
  const unsigned long long out_a    = (unsigned long long)out;

  for (int i = tid; i < 16 * KTOT; i += 512) ((__bf16*)h_lds)[i] = (__bf16)0.0f;
  if (tid < NTGT) blin_s[tid] = b_lin[tid];

  // per-wave constants (hoisted out of the T loop)
  float biasr[4][2];
#pragma unroll
  for (int g = 0; g < 4; ++g)
#pragma unroll
    for (int q = 0; q < 2; ++q)
      biasr[g][q] = bias_c[g * NH + w * 32 + q * 16 + ln16];

  const v16bf wlinB =
      ld_frag_gaddr((unsigned long long)WlRaw +
                    (unsigned long long)(w * 32 + lane) * 32ull);

  v8f cst[2];
#pragma unroll
  for (int q = 0; q < 2; ++q)
#pragma unroll
    for (int r = 0; r < 8; ++r) cst[q][r] = 0.0f;

  __syncthreads();

  for (int t = 0; t < T_STEPS; ++t) {
    // ---- stage x_t (bf16) into K columns [512, 546); [546,576) stays zero
    for (int i = tid; i < 16 * NPTS; i += 512) {
      int r = i / NPTS, ii = i - r * NPTS;
      float xv = *(gcfloatp)(points_a +
                 (((unsigned long long)t * BATCH + bbase + r) * NPTS + ii) * 4ull);
      h_lds[r][NH + ii] = (__bf16)xv;
    }
    __syncthreads();

    // Launder the weight base (SGPR) each step: loads stay in the t-loop.
    unsigned long long wcb = (unsigned long long)WcRaw;
    asm volatile("" : "+s"(wcb));

    // per-(gate,q) tile base addresses; imm offset inside the c-loop = c*1KB
    unsigned long long tba[8];
#pragma unroll
    for (int g = 0; g < 4; ++g)
#pragma unroll
      for (int q = 0; q < 2; ++q)
        tba[g * 2 + q] = wcb +
            (unsigned long long)(g * 32 + 2 * w + q) * TILE_BYTES +
            (unsigned long long)(lane * 32);

    // ---- gates = [h | x] * Wc^T + bias  (bf16 WMMA, f32 accumulate)
    v8f acc[4][2];
#pragma unroll
    for (int g = 0; g < 4; ++g)
#pragma unroll
      for (int q = 0; q < 2; ++q)
#pragma unroll
        for (int r = 0; r < 8; ++r) acc[g][q][r] = biasr[g][q];

#pragma unroll
    for (int c = 0; c < KCH; ++c) {
      // A fragment for this K-chunk from LDS (h | x)
      const int k0 = c * 32 + grp * 16;
      FragU af;
      af.u[0] = *(const u32x4*)&h_lds[ln16][k0];
      af.u[1] = *(const u32x4*)&h_lds[ln16][k0 + 8];
      const v16bf a = af.v;

      // stage all 8 B fragments for this chunk, then issue the 8 WMMAs:
      // bounded one-chunk pipeline window for the scheduler.
      v16bf bfr[8];
#pragma unroll
      for (int j = 0; j < 8; ++j)
        bfr[j] = ld_frag_gaddr(tba[j] + (unsigned)(c * 1024));

#pragma unroll
      for (int g = 0; g < 4; ++g)
#pragma unroll
        for (int q = 0; q < 2; ++q)
          acc[g][q] = __builtin_amdgcn_wmma_f32_16x16x32_bf16(
              false, a, false, bfr[g * 2 + q], (short)0, acc[g][q],
              false, false);
    }

    // ---- elementwise gate math; i/f/g/o share the same C-layout (M,N) map
    v8f hval[2];
#pragma unroll
    for (int q = 0; q < 2; ++q) {
#pragma unroll
      for (int r = 0; r < 8; ++r) {
        float iv = 1.0f / (1.0f + __expf(-acc[0][q][r]));
        float fv = 1.0f / (1.0f + __expf(-acc[1][q][r]));
        float gv = tanhf(acc[2][q][r]);
        float ov = 1.0f / (1.0f + __expf(-acc[3][q][r]));
        float cv = fv * cst[q][r] + iv * gv;
        cst[q][r]  = cv;
        hval[q][r] = ov * tanhf(cv);
      }
    }

    __syncthreads();   // all waves finished reading h_{t-1}

#pragma unroll
    for (int q = 0; q < 2; ++q)
#pragma unroll
      for (int r = 0; r < 8; ++r) {
        int M = r + grp * 8;                       // C-layout row
        h_lds[M][w * 32 + q * 16 + ln16] = (__bf16)hval[q][r];
      }

    __syncthreads();   // h_t visible to all waves

    // ---- fused output head: wave w does K-chunk w of logits = h * W_lin^T
    {
      const int k0 = w * 32 + grp * 16;
      FragU af;
      af.u[0] = *(const u32x4*)&h_lds[ln16][k0];
      af.u[1] = *(const u32x4*)&h_lds[ln16][k0 + 8];
      v8f lg;
#pragma unroll
      for (int r = 0; r < 8; ++r) lg[r] = 0.0f;
      lg = __builtin_amdgcn_wmma_f32_16x16x32_bf16(
          false, af.v, false, wlinB, (short)0, lg, false, false);
#pragma unroll
      for (int r = 0; r < 8; ++r)
        lpart[w * 256 + (r + grp * 8) * 16 + ln16] = lg[r];
    }
    __syncthreads();

    // deterministic LDS reduce of the 16 partial tiles, summed serially in wave order
    if (tid < 256) {
      float s = 0.0f;
#pragma unroll
      for (int ww = 0; ww < 16; ++ww) s += lpart[ww * 256 + tid];
      lred[tid] = s;
    }
    __syncthreads();

    // softmax per batch row, store to out[t*B + bbase + row][0..15]
    if (tid < 16) {
      float lv[NTGT], m = -3.0e38f;
#pragma unroll
      for (int n = 0; n < NTGT; ++n) {
        lv[n] = lred[tid * 16 + n] + blin_s[n];
        m = fmaxf(m, lv[n]);
      }
      float s = 0.0f;
#pragma unroll
      for (int n = 0; n < NTGT; ++n) { lv[n] = __expf(lv[n] - m); s += lv[n]; }
      float inv = 1.0f / s;
      unsigned long long ob = out_a +
          (((unsigned long long)t * BATCH + bbase + tid) * NTGT) * 4ull;
#pragma unroll
      for (int n = 0; n < NTGT; ++n)
        *(gfloatp)(ob + n * 4ull) = lv[n] * inv;
    }
    // no trailing barrier needed: next-iteration writers are all separated
    // from this iteration's readers by at least one __syncthreads above.
  }
}

extern "C" void kernel_launch(void* const* d_in, const int* in_sizes, int n_in,
                              void* d_out, int out_size, void* d_ws, size_t ws_size,
                              hipStream_t stream) {
  const float* points = (const float*)d_in[0];
  const float* W_ih   = (const float*)d_in[1];
  const float* W_hh   = (const float*)d_in[2];
  const float* b_ih   = (const float*)d_in[3];
  const float* b_hh   = (const float*)d_in[4];
  const float* W_lin  = (const float*)d_in[5];
  const float* b_lin  = (const float*)d_in[6];

  char* ws = (char*)d_ws;
  unsigned short* WcRaw = (unsigned short*)(ws);                       // 2.25 MB bf16
  unsigned short* WlRaw = (unsigned short*)(ws + WC_BYTES);            // 16 KB bf16
  float*          bc    = (float*)(ws + WC_BYTES + WL_BYTES);          // 8 KB f32

  pack_wc_kernel  <<<(WC_ELEMS + 255) / 256, 256, 0, stream>>>(W_ih, W_hh, WcRaw);
  pack_wlin_kernel<<<(WL_ELEMS + 255) / 256, 256, 0, stream>>>(W_lin, WlRaw);
  bias_kernel     <<<(NG + 255) / 256, 256, 0, stream>>>(b_ih, b_hh, bc);

  lstm_persist_kernel<<<BATCH / 16, 512, 0, stream>>>(
      points, WcRaw, WlRaw, bc, b_lin, (float*)d_out);
}
